// RegisterTokenTransformer_42545946034270
// MI455X (gfx1250) — matrix operand order repaired and checked by reading, not compiled, not run on hardware
//
#include <hip/hip_runtime.h>
#include <cstddef>

// ---------------------------------------------------------------- types
typedef __bf16 bf16_t;
typedef __attribute__((ext_vector_type(4)))  __bf16 v4bf;
typedef __attribute__((ext_vector_type(8)))  __bf16 v8bf;
typedef __attribute__((ext_vector_type(16))) __bf16 v16bf;
typedef __attribute__((ext_vector_type(8)))  float  v8f;

#define DEV __device__ __forceinline__

DEV v8f zero8() {
  v8f z;
#pragma unroll
  for (int i = 0; i < 8; ++i) z[i] = 0.f;
  return z;
}

DEV v16bf cat16(v8bf lo, v8bf hi) {
  return __builtin_shufflevector(lo, hi, 0,1,2,3,4,5,6,7,8,9,10,11,12,13,14,15);
}

// A-matrix fragment (16x32 bf16): per lane contiguous 16 values at sel*16
DEV v16bf ld_a_frag(const bf16_t* rowbase, int sel) {
  const v8bf* lo = (const v8bf*)(rowbase + sel * 16);
  const v8bf* hi = (const v8bf*)(rowbase + sel * 16 + 8);
  return cat16(*lo, *hi);
}
// B-matrix fragment (32x16 bf16): per lane contiguous 16 values at sel*16
DEV v16bf ld_b_frag(const bf16_t* rowbase, int sel) {
  const v8bf* p = (const v8bf*)(rowbase + sel * 16);
  return cat16(p[0], p[1]);
}

DEV v8f wmma_bf16(v16bf a, v16bf b, v8f c) {
  return __builtin_amdgcn_wmma_f32_16x16x32_bf16(
      false, a, false, b, (short)0, c, false, false);
}

// ---------------------------------------------------------------- config
static constexpr int BB   = 8;
static constexpr int SS   = 1040;   // 1024 inputs + 16 registers
static constexpr int NINP = 1024;
static constexpr int DD   = 1024;
static constexpr int HH   = 16;
static constexpr int HDIM = 64;
static constexpr int LL   = 12;
static constexpr int DFFN = 4096;
static constexpr int MM   = BB * SS;   // 8320 = 65*128

// ---------------------------------------------------------------- layernorm
template <bool OUTF32>
__global__ void __launch_bounds__(256)
ln_kernel(const float* __restrict__ x, const float* __restrict__ g,
          const float* __restrict__ bta, void* __restrict__ outv) {
  const int row = blockIdx.x, tid = threadIdx.x;
  const float* xr = x + (size_t)row * DD;
  float v[4], s = 0.f, s2 = 0.f;
#pragma unroll
  for (int i = 0; i < 4; ++i) {
    v[i] = xr[tid + i * 256];
    s += v[i]; s2 += v[i] * v[i];
  }
#pragma unroll
  for (int off = 16; off; off >>= 1) {
    s  += __shfl_xor(s,  off, 32);
    s2 += __shfl_xor(s2, off, 32);
  }
  __shared__ float r1[8], r2[8];
  const int wave = tid >> 5, lane = tid & 31;
  if (lane == 0) { r1[wave] = s; r2[wave] = s2; }
  __syncthreads();
  if (wave == 0) {
    float a = (lane < 8) ? r1[lane] : 0.f;
    float b = (lane < 8) ? r2[lane] : 0.f;
#pragma unroll
    for (int off = 4; off; off >>= 1) {
      a += __shfl_xor(a, off, 32);
      b += __shfl_xor(b, off, 32);
    }
    if (lane == 0) { r1[0] = a; r2[0] = b; }
  }
  __syncthreads();
  const float mu  = r1[0] * (1.f / DD);
  const float var = r2[0] * (1.f / DD) - mu * mu;
  const float rs  = rsqrtf(var + 1e-6f);
#pragma unroll
  for (int i = 0; i < 4; ++i) {
    const int idx = tid + i * 256;
    const float y = (v[i] - mu) * rs * g[idx] + bta[idx];
    if (OUTF32) ((float*)outv)[(size_t)row * DD + idx] = y;
    else        ((bf16_t*)outv)[(size_t)row * DD + idx] = (bf16_t)y;
  }
}

// ---------------------------------------------------------------- f32 -> bf16
__global__ void __launch_bounds__(256)
cvt_kernel(const float* __restrict__ in, bf16_t* __restrict__ out, int n4) {
  const int i = blockIdx.x * 256 + threadIdx.x;
  if (i < n4) {
    const float4 v = ((const float4*)in)[i];
    v4bf o; o[0] = (bf16_t)v.x; o[1] = (bf16_t)v.y;
            o[2] = (bf16_t)v.z; o[3] = (bf16_t)v.w;
    *(v4bf*)(out + (size_t)i * 4) = o;
  }
}

// ---------------------------------------------------------------- GEMM
// C(M,N) = A(M,K) @ W(K,N) + bias.  Block tile 128x128, BK=32, 8 waves,
// each wave 32x64 (2x4 WMMA tiles).  Global->LDS staging is software-
// pipelined through registers.  EPI: 0=store bf16, 1=+= into f32, 2=gelu->bf16
template <int EPI>
__global__ void __launch_bounds__(256)
gemm_kernel(const bf16_t* __restrict__ A, const bf16_t* __restrict__ W,
            const float* __restrict__ bias, float* __restrict__ Cf,
            bf16_t* __restrict__ Cb, int M, int N, int K) {
  __shared__ bf16_t As[128 * 48];    // [m][k]
  __shared__ bf16_t Bs[128 * 48];    // transposed: [n][k]
  const int tid = threadIdx.x, lane = tid & 31, wave = tid >> 5;
  const int sel = lane >> 4, l15 = lane & 15;
  const int bm = blockIdx.y * 128, bn = blockIdx.x * 128;
  const int wm = (wave >> 1) * 32, wn = (wave & 1) * 64;

  v8f acc[2][4];
#pragma unroll
  for (int i = 0; i < 2; ++i)
#pragma unroll
    for (int j = 0; j < 4; ++j) acc[i][j] = zero8();

  v8bf aReg[2], wReg[2];
  // A tile 128x32 = 512 chunks of 8; W tile 32x128 = 512 chunks of 8
  auto gload = [&](int k0) {
#pragma unroll
    for (int it = 0; it < 2; ++it) {
      const int c = tid + it * 256;
      aReg[it] = *(const v8bf*)(A + (size_t)(bm + (c >> 2)) * K + k0 + (c & 3) * 8);
      wReg[it] = *(const v8bf*)(W + (size_t)(k0 + (c >> 4)) * N + bn + (c & 15) * 8);
    }
  };
  auto sstore = [&]() {
#pragma unroll
    for (int it = 0; it < 2; ++it) {
      const int c = tid + it * 256;
      *(v8bf*)(As + (c >> 2) * 48 + (c & 3) * 8) = aReg[it];
      const int kk = c >> 4, n0 = (c & 15) * 8;
#pragma unroll
      for (int j = 0; j < 8; ++j) Bs[(n0 + j) * 48 + kk] = wReg[it][j];
    }
  };

  gload(0);
  for (int k0 = 0; k0 < K; k0 += 32) {
    sstore();
    __syncthreads();
    if (k0 + 32 < K) gload(k0 + 32);   // prefetch next tile while computing

    v16bf afr[2], bfr[4];
#pragma unroll
    for (int s = 0; s < 2; ++s)
      afr[s] = ld_a_frag(As + (wm + s * 16 + l15) * 48, sel);
#pragma unroll
    for (int s = 0; s < 4; ++s)
      bfr[s] = ld_b_frag(Bs + (wn + s * 16 + l15) * 48, sel);
#pragma unroll
    for (int i = 0; i < 2; ++i)
#pragma unroll
      for (int j = 0; j < 4; ++j)
        acc[i][j] = wmma_bf16(afr[i], bfr[j], acc[i][j]);
    __syncthreads();
  }

#pragma unroll
  for (int i = 0; i < 2; ++i)
#pragma unroll
    for (int j = 0; j < 4; ++j) {
      const int ng = bn + wn + j * 16 + l15;
      const float bia = bias[ng];
#pragma unroll
      for (int r = 0; r < 8; ++r) {
        const int mg = bm + wm + i * 16 + r + 8 * sel;
        const float v = acc[i][j][r] + bia;
        if (EPI == 0) {
          Cb[(size_t)mg * N + ng] = (bf16_t)v;
        } else if (EPI == 1) {
          Cf[(size_t)mg * N + ng] += v;
        } else {
          const float t = 0.7978845608f * (v + 0.044715f * v * v * v);
          Cb[(size_t)mg * N + ng] = (bf16_t)(0.5f * v * (1.f + tanhf(t)));
        }
      }
    }
}

// ---------------------------------------------------------------- attention
// qkv: (B*S, 3D) bf16 row-major.  out: (B*S, D) bf16.
// Block = 5 waves = 80 query rows (1040 = 13*80); key tiles of 32.
__global__ void __launch_bounds__(160)
attn_kernel(const bf16_t* __restrict__ qkv, bf16_t* __restrict__ out) {
  __shared__ bf16_t Ks[32 * 72];        // [key][dim]
  __shared__ bf16_t Vt[64 * 48];        // [dim][key]
  __shared__ bf16_t Ps[5][16 * 48];     // wave-private P tiles
  const int tid = threadIdx.x, lane = tid & 31, wave = tid >> 5;
  const int sel = lane >> 4, l15 = lane & 15;
  const int qt = blockIdx.x % 13;
  const int bh = blockIdx.x / 13;
  const int b = bh >> 4, h = bh & 15;

  const int qrow = qt * 80 + wave * 16 + l15;         // 0..1039
  const bf16_t* qp = qkv + ((size_t)b * SS + qrow) * (3 * DD) + h * HDIM;
  v16bf qf[2];
#pragma unroll
  for (int c = 0; c < 2; ++c) qf[c] = ld_a_frag(qp + c * 32, sel);

  v8f o[4]; float mrow[8], lrow[8];
#pragma unroll
  for (int n = 0; n < 4; ++n) o[n] = zero8();
#pragma unroll
  for (int r = 0; r < 8; ++r) { mrow[r] = -1e30f; lrow[r] = 0.f; }

  for (int kt = 0; kt < 33; ++kt) {
    const int kbase = kt * 32;
    __syncthreads();
    // cooperative K / V(transposed) tile load: 256 chunks of 8
    for (int c = tid; c < 256; c += 160) {
      const int key = c >> 3, d0 = (c & 7) * 8;
      const int kg = kbase + key;
      v8bf kd, vd;
      if (kg < SS) {
        const bf16_t* base = qkv + ((size_t)b * SS + kg) * (3 * DD) + h * HDIM + d0;
        kd = *(const v8bf*)(base + DD);
        vd = *(const v8bf*)(base + 2 * DD);
      } else {
#pragma unroll
        for (int j = 0; j < 8; ++j) { kd[j] = (bf16_t)0.f; vd[j] = (bf16_t)0.f; }
      }
      *(v8bf*)(Ks + key * 72 + d0) = kd;
#pragma unroll
      for (int j = 0; j < 8; ++j) Vt[(d0 + j) * 48 + key] = vd[j];
    }
    __syncthreads();

    // scores: Q(16x64) @ K^T -> two 16x16 C tiles
    v8f sc[2];
#pragma unroll
    for (int ns = 0; ns < 2; ++ns) {
      sc[ns] = zero8();
#pragma unroll
      for (int c = 0; c < 2; ++c) {
        v16bf kf = ld_b_frag(Ks + (ns * 16 + l15) * 72 + c * 32, sel);
        sc[ns] = wmma_bf16(qf[c], kf, sc[ns]);
      }
    }

    // online softmax (rows live per C-tile vgpr: row = r + 8*sel)
#pragma unroll
    for (int r = 0; r < 8; ++r) {
      const int qg = qt * 80 + wave * 16 + r + 8 * sel;
#pragma unroll
      for (int ns = 0; ns < 2; ++ns) {
        const int kg = kbase + ns * 16 + l15;
        float s = sc[ns][r] * 0.125f;
        const bool blocked = (kg >= SS) || (qg >= NINP && kg >= NINP && kg > qg);
        sc[ns][r] = blocked ? -1e30f : s;
      }
      float mx = fmaxf(sc[0][r], sc[1][r]);
#pragma unroll
      for (int off = 8; off; off >>= 1) mx = fmaxf(mx, __shfl_xor(mx, off, 32));
      const float mnew = fmaxf(mrow[r], mx);
      float sum = 0.f;
#pragma unroll
      for (int ns = 0; ns < 2; ++ns) {
        const float e = __expf(sc[ns][r] - mnew);
        sc[ns][r] = e; sum += e;
      }
#pragma unroll
      for (int off = 8; off; off >>= 1) sum += __shfl_xor(sum, off, 32);
      const float alpha = __expf(mrow[r] - mnew);
      lrow[r] = lrow[r] * alpha + sum;
      mrow[r] = mnew;
#pragma unroll
      for (int n = 0; n < 4; ++n) o[n][r] *= alpha;
    }

    // C-layout -> A-layout for P via wave-private LDS
#pragma unroll
    for (int ns = 0; ns < 2; ++ns)
#pragma unroll
      for (int r = 0; r < 8; ++r)
        Ps[wave][(r + 8 * sel) * 48 + ns * 16 + l15] = (bf16_t)sc[ns][r];
    asm volatile("s_wait_dscnt 0x0" ::: "memory");
    v16bf pf = ld_a_frag(&Ps[wave][l15 * 48], sel);

    // O += P(16x32) @ V(32x64)
#pragma unroll
    for (int ns = 0; ns < 4; ++ns) {
      v16bf vf = ld_b_frag(Vt + (ns * 16 + l15) * 48, sel);
      o[ns] = wmma_bf16(pf, vf, o[ns]);
    }
  }

  // normalize + write
  float inv[8];
#pragma unroll
  for (int r = 0; r < 8; ++r) inv[r] = 1.f / lrow[r];
#pragma unroll
  for (int ns = 0; ns < 4; ++ns)
#pragma unroll
    for (int r = 0; r < 8; ++r) {
      const int row = qt * 80 + wave * 16 + r + 8 * sel;
      const int col = h * HDIM + ns * 16 + l15;
      out[((size_t)b * SS + row) * DD + col] = (bf16_t)(o[ns][r] * inv[r]);
    }
}

// ---------------------------------------------------------------- host
extern "C" void kernel_launch(void* const* d_in, const int* in_sizes, int n_in,
                              void* d_out, int out_size, void* d_ws, size_t ws_size,
                              hipStream_t stream) {
  (void)in_sizes; (void)n_in; (void)out_size; (void)ws_size;
  const float* x     = (const float*)d_in[0];
  const float* ln1_g = (const float*)d_in[2];
  const float* ln1_b = (const float*)d_in[3];
  const float* wqkv  = (const float*)d_in[4];
  const float* bqkv  = (const float*)d_in[5];
  const float* wo    = (const float*)d_in[6];
  const float* bo    = (const float*)d_in[7];
  const float* ln2_g = (const float*)d_in[8];
  const float* ln2_b = (const float*)d_in[9];
  const float* w1    = (const float*)d_in[10];
  const float* b1    = (const float*)d_in[11];
  const float* w2    = (const float*)d_in[12];
  const float* b2    = (const float*)d_in[13];
  const float* lnf_g = (const float*)d_in[14];
  const float* lnf_b = (const float*)d_in[15];

  char* base = (char*)d_ws;
  size_t off = 0;
  auto carve = [&](size_t bytes) -> void* {
    void* p = base + off;
    off += (bytes + 255) & ~(size_t)255;
    return p;
  };
  float*  h_f    = (float*) carve((size_t)MM * DD * 4);
  bf16_t* ln_bf  = (bf16_t*)carve((size_t)MM * DD * 2);
  bf16_t* qkv_bf = (bf16_t*)carve((size_t)MM * 3 * DD * 2);
  bf16_t* att_bf = (bf16_t*)carve((size_t)MM * DD * 2);
  bf16_t* ffn_bf = (bf16_t*)carve((size_t)MM * DFFN * 2);
  bf16_t* wqkv_bf= (bf16_t*)carve((size_t)DD * 3 * DD * 2);
  bf16_t* wo_bf  = (bf16_t*)carve((size_t)DD * DD * 2);
  bf16_t* w1_bf  = (bf16_t*)carve((size_t)DD * DFFN * 2);
  bf16_t* w2_bf  = (bf16_t*)carve((size_t)DFFN * DD * 2);

  (void)hipMemcpyAsync(h_f, x, (size_t)MM * DD * 4, hipMemcpyDeviceToDevice, stream);

  auto cvt = [&](const float* src, bf16_t* dst, size_t n) {
    const int n4 = (int)(n / 4);
    cvt_kernel<<<(n4 + 255) / 256, 256, 0, stream>>>(src, dst, n4);
  };

  const dim3 blk(256);
  for (int l = 0; l < LL; ++l) {
    const size_t oD = (size_t)l * DD;
    // --- attention sub-block
    ln_kernel<false><<<MM, blk, 0, stream>>>(h_f, ln1_g + oD, ln1_b + oD, ln_bf);
    cvt(wqkv + (size_t)l * DD * 3 * DD, wqkv_bf, (size_t)DD * 3 * DD);
    gemm_kernel<0><<<dim3(3 * DD / 128, MM / 128), blk, 0, stream>>>(
        ln_bf, wqkv_bf, bqkv + (size_t)l * 3 * DD, nullptr, qkv_bf, MM, 3 * DD, DD);
    attn_kernel<<<BB * HH * 13, dim3(160), 0, stream>>>(qkv_bf, att_bf);
    cvt(wo + (size_t)l * DD * DD, wo_bf, (size_t)DD * DD);
    gemm_kernel<1><<<dim3(DD / 128, MM / 128), blk, 0, stream>>>(
        att_bf, wo_bf, bo + oD, h_f, nullptr, MM, DD, DD);
    // --- MLP sub-block
    ln_kernel<false><<<MM, blk, 0, stream>>>(h_f, ln2_g + oD, ln2_b + oD, ln_bf);
    cvt(w1 + (size_t)l * DD * DFFN, w1_bf, (size_t)DD * DFFN);
    gemm_kernel<2><<<dim3(DFFN / 128, MM / 128), blk, 0, stream>>>(
        ln_bf, w1_bf, b1 + (size_t)l * DFFN, nullptr, ffn_bf, MM, DFFN, DD);
    cvt(w2 + (size_t)l * DFFN * DD, w2_bf, (size_t)DFFN * DD);
    gemm_kernel<1><<<dim3(DD / 128, MM / 128), blk, 0, stream>>>(
        ffn_bf, w2_bf, b2 + oD, h_f, nullptr, MM, DD, DFFN);
  }
  ln_kernel<true><<<MM, blk, 0, stream>>>(h_f, lnf_g, lnf_b, (float*)d_out);
}
